// AttentionFusion_18897856102556
// MI455X (gfx1250) — matrix operand order repaired and checked by reading, not compiled, not run on hardware
//
#include <hip/hip_runtime.h>
#include <stdint.h>

// ---------------------------------------------------------------------------
// CDNA5 WMMA types / helpers
// ---------------------------------------------------------------------------
typedef __attribute__((ext_vector_type(16))) __bf16        v16bf;
typedef __attribute__((ext_vector_type(8)))  float         v8f;
typedef __attribute__((ext_vector_type(4)))  unsigned int  v4u;

union Frag16 { v16bf v; v4u q[2]; };

__device__ __forceinline__ unsigned short f2bf(float x) {
    union { float f; uint32_t u; } c; c.f = x;
    uint32_t r = (c.u + 0x7FFFu + ((c.u >> 16) & 1u)) >> 16;  // RNE
    return (unsigned short)r;
}

#define WMMA_BF16(A, B, C) \
    __builtin_amdgcn_wmma_f32_16x16x32_bf16(false, (A), false, (B), (short)0, (C), false, false)

// ---------------------------------------------------------------------------
// Elementwise: f32 -> bf16 cast
// ---------------------------------------------------------------------------
__global__ void cast_bf16(const float* __restrict__ s, unsigned short* __restrict__ d, int n) {
    int i = blockIdx.x * blockDim.x + threadIdx.x;
    if (i < n) d[i] = f2bf(s[i]);
}

// ---------------------------------------------------------------------------
// Point projection: pf = points @ Wp.T + bp  (K = 3, trivial; bf16 out into
// the concat buffer Xf[:, 0:256], row stride 512)
// ---------------------------------------------------------------------------
__global__ void point_proj(const float* __restrict__ pts, const float* __restrict__ Wp,
                           const float* __restrict__ bp, unsigned short* __restrict__ Xf,
                           int ntok) {
    int i = blockIdx.x * blockDim.x + threadIdx.x;
    int o = i & 255;
    int t = i >> 8;
    if (t >= ntok) return;
    float x0 = pts[t * 3 + 0], x1 = pts[t * 3 + 1], x2 = pts[t * 3 + 2];
    float v = bp[o] + x0 * Wp[o * 3 + 0] + x1 * Wp[o * 3 + 1] + x2 * Wp[o * 3 + 2];
    Xf[(size_t)t * 512 + o] = f2bf(v);
}

// ---------------------------------------------------------------------------
// WMMA GEMM: out = (X[rows,K](bf16, stride ldx) @ W[OUT,K].T + bias) * scale
// One wave per 16x64 output tile (A-fragment reuse across 4 B-fragments).
// MODE 0: bf16 row-major out    MODE 1: f32 row-major out
// MODE 2: bf16 out in vT layout [(b*256 + col) * Mv + m]  (row = b*Mv + m)
// ---------------------------------------------------------------------------
template <int MODE>
__global__ __launch_bounds__(32) void wmma_gemm(
    const unsigned short* __restrict__ X, int ldx,
    const unsigned short* __restrict__ W, const float* __restrict__ bias,
    void* __restrict__ outp, int K, int OUT, float scale, int Mv) {
    const int lane = threadIdx.x;
    const int arow = lane & 15, ahalf = lane >> 4;
    const int rowbase = blockIdx.x * 16;
    const int obase = blockIdx.y * 64;

    v8f acc[4] = {};
    const unsigned short* xrow = X + (size_t)(rowbase + arow) * ldx + ahalf * 8;
    for (int kb = 0; kb < K; kb += 32) {
        Frag16 a;                                   // A 16x32: K = kbase + {0..7, 16..23}
        a.q[0] = *(const v4u*)(xrow + kb);
        a.q[1] = *(const v4u*)(xrow + kb + 16);
#pragma unroll
        for (int t = 0; t < 4; ++t) {
            const unsigned short* wp =
                W + (size_t)(obase + t * 16 + arow) * K + kb + ahalf * 8;
            Frag16 bfr;                             // B 32x16: lane=col, K = kbase + {0..7, 16..23}
            bfr.q[0] = *(const v4u*)(wp);
            bfr.q[1] = *(const v4u*)(wp + 16);
            acc[t] = WMMA_BF16(a.v, bfr.v, acc[t]);
        }
    }
#pragma unroll
    for (int t = 0; t < 4; ++t) {
        int col = obase + t * 16 + arow;
        float bv = bias[col];
#pragma unroll
        for (int r = 0; r < 8; ++r) {
            int row = rowbase + r + ahalf * 8;      // D layout: lane=col, rows striped
            float val = (acc[t][r] + bv) * scale;
            if (MODE == 0) {
                ((unsigned short*)outp)[(size_t)row * OUT + col] = f2bf(val);
            } else if (MODE == 1) {
                ((float*)outp)[(size_t)row * OUT + col] = val;
            } else {
                int bb = row / Mv, mm = row % Mv;
                ((unsigned short*)outp)[((size_t)bb * 256 + col) * Mv + mm] = f2bf(val);
            }
        }
    }
}

// ---------------------------------------------------------------------------
// Flash attention: one wave per (batch, head, 16-query tile).
// q pre-scaled by 1/sqrt(HD). Iterates keys in blocks of 32:
//   S (2x WMMA QK^T) -> LDS -> online softmax -> P bf16 -> 2x WMMA PV.
// Writes attended (bf16) into Xf[:, 256:512] for the fusion GEMM.
// ---------------------------------------------------------------------------
__global__ __launch_bounds__(32) void flash_attn(
    const unsigned short* __restrict__ qb, const unsigned short* __restrict__ kbuf,
    const unsigned short* __restrict__ vT, unsigned short* __restrict__ Xf,
    int N, int M) {
    __shared__ float sS[16 * 32];
    __shared__ __align__(16) unsigned short pS[16 * 32];
    __shared__ float aS[16];
    __shared__ float lS[16];

    const int lane = threadIdx.x;
    const int arow = lane & 15, ahalf = lane >> 4;

    const int tiles = N >> 4;
    const int qt = blockIdx.x % tiles;
    const int h = (blockIdx.x / tiles) & 7;
    const int b = blockIdx.x / (tiles * 8);

    Frag16 aQ;  // Q tile 16x32 (head_dim) — one WMMA K-step
    const unsigned short* qrow =
        qb + (size_t)(b * N + qt * 16 + arow) * 256 + h * 32 + ahalf * 8;
    aQ.q[0] = *(const v4u*)(qrow);
    aQ.q[1] = *(const v4u*)(qrow + 16);

    v8f acc0 = {}, acc1 = {};
    float m_i = -1e30f, l_i = 0.0f;

    const unsigned short* vbase0 = vT + (size_t)((b * 8 + h) * 32 + arow) * M;
    const unsigned short* vbase1 = vbase0 + (size_t)16 * M;

    for (int kb0 = 0; kb0 < M; kb0 += 32) {
#pragma unroll
        for (int f = 0; f < 2; ++f) {
            const unsigned short* kp =
                kbuf + (size_t)(b * M + kb0 + f * 16 + arow) * 256 + h * 32 + ahalf * 8;
            Frag16 bK;
            bK.q[0] = *(const v4u*)(kp);
            bK.q[1] = *(const v4u*)(kp + 16);
            v8f zero = {};
            v8f s = WMMA_BF16(aQ.v, bK.v, zero);
#pragma unroll
            for (int r = 0; r < 8; ++r)
                sS[(r + ahalf * 8) * 32 + f * 16 + arow] = s[r];
        }
        __syncthreads();
        if (lane < 16) {  // one lane per query row: online softmax
            float mx = m_i;
#pragma unroll
            for (int j = 0; j < 32; ++j) mx = fmaxf(mx, sS[lane * 32 + j]);
            float alpha = __expf(m_i - mx);
            float sum = 0.0f;
#pragma unroll
            for (int j = 0; j < 32; ++j) {
                float p = __expf(sS[lane * 32 + j] - mx);
                sum += p;
                pS[lane * 32 + j] = f2bf(p);
            }
            l_i = l_i * alpha + sum;
            m_i = mx;
            aS[lane] = alpha;
        }
        __syncthreads();
#pragma unroll
        for (int r = 0; r < 8; ++r) {
            float al = aS[r + ahalf * 8];
            acc0[r] *= al;
            acc1[r] *= al;
        }
        Frag16 aP;  // P 16x32 in A layout, straight from LDS (16B ds_load_b128)
        const unsigned short* pp = pS + arow * 32 + ahalf * 8;
        aP.q[0] = *(const v4u*)(pp);
        aP.q[1] = *(const v4u*)(pp + 16);
        Frag16 bV0, bV1;  // V^T: keys {0..7, 16..23} + 8*half per lane
        const unsigned short* vp0 = vbase0 + kb0 + ahalf * 8;
        const unsigned short* vp1 = vbase1 + kb0 + ahalf * 8;
        bV0.q[0] = *(const v4u*)(vp0);
        bV0.q[1] = *(const v4u*)(vp0 + 16);
        bV1.q[0] = *(const v4u*)(vp1);
        bV1.q[1] = *(const v4u*)(vp1 + 16);
        acc0 = WMMA_BF16(aP.v, bV0.v, acc0);
        acc1 = WMMA_BF16(aP.v, bV1.v, acc1);
        __syncthreads();
    }
    if (lane < 16) lS[lane] = l_i;
    __syncthreads();
#pragma unroll
    for (int r = 0; r < 8; ++r) {
        int row = r + ahalf * 8;
        float inv = 1.0f / lS[row];
        size_t tok = (size_t)(b * N + qt * 16 + row);
        Xf[tok * 512 + 256 + h * 32 + arow]      = f2bf(acc0[r] * inv);
        Xf[tok * 512 + 256 + h * 32 + 16 + arow] = f2bf(acc1[r] * inv);
    }
}

// ---------------------------------------------------------------------------
// Host pipeline
// ---------------------------------------------------------------------------
extern "C" void kernel_launch(void* const* d_in, const int* in_sizes, int n_in,
                              void* d_out, int out_size, void* d_ws, size_t ws_size,
                              hipStream_t stream) {
    const float* points = (const float*)d_in[0];
    const float* voxel  = (const float*)d_in[1];
    const float* Wp  = (const float*)d_in[2];
    const float* bp  = (const float*)d_in[3];
    const float* Wq  = (const float*)d_in[4];
    const float* bq  = (const float*)d_in[5];
    const float* Wk  = (const float*)d_in[6];
    const float* bk  = (const float*)d_in[7];
    const float* Wv  = (const float*)d_in[8];
    const float* bv  = (const float*)d_in[9];
    const float* Wf  = (const float*)d_in[10];
    const float* bfp = (const float*)d_in[11];

    const int B = 2, N = 4096, M = 2048;
    const int BN = B * N;   // 8192 rows (mult of 16)
    const int BM = B * M;   // 4096 rows (mult of 16)

    char* w = (char*)d_ws;
    auto alloc = [&](size_t bytes) {
        char* p = w;
        w += (bytes + 255) & ~(size_t)255;
        return p;
    };
    unsigned short* Xf   = (unsigned short*)alloc((size_t)BN * 512 * 2);  // [pf | attended] bf16
    unsigned short* qb   = (unsigned short*)alloc((size_t)BN * 256 * 2);
    unsigned short* kb   = (unsigned short*)alloc((size_t)BM * 256 * 2);
    unsigned short* vT   = (unsigned short*)alloc((size_t)B * 256 * M * 2);  // [b,h,d,m]
    unsigned short* voxb = (unsigned short*)alloc((size_t)BM * 256 * 2);
    unsigned short* Wqb  = (unsigned short*)alloc((size_t)256 * 256 * 2);
    unsigned short* Wkb  = (unsigned short*)alloc((size_t)256 * 256 * 2);
    unsigned short* Wvb  = (unsigned short*)alloc((size_t)256 * 256 * 2);
    unsigned short* Wfb  = (unsigned short*)alloc((size_t)256 * 512 * 2);

    int n;
    n = BM * 256;  cast_bf16<<<(n + 255) / 256, 256, 0, stream>>>(voxel, voxb, n);
    n = 256 * 256; cast_bf16<<<(n + 255) / 256, 256, 0, stream>>>(Wq, Wqb, n);
    n = 256 * 256; cast_bf16<<<(n + 255) / 256, 256, 0, stream>>>(Wk, Wkb, n);
    n = 256 * 256; cast_bf16<<<(n + 255) / 256, 256, 0, stream>>>(Wv, Wvb, n);
    n = 256 * 512; cast_bf16<<<(n + 255) / 256, 256, 0, stream>>>(Wf, Wfb, n);

    n = BN * 256;
    point_proj<<<(n + 255) / 256, 256, 0, stream>>>(points, Wp, bp, Xf, BN);

    const float scale = 0.17677669529663687f;  // 1/sqrt(32), folded into q
    wmma_gemm<0><<<dim3(BN / 16, 256 / 64), 32, 0, stream>>>(Xf, 512, Wqb, bq, qb, 256, 256, scale, 1);
    wmma_gemm<0><<<dim3(BM / 16, 256 / 64), 32, 0, stream>>>(voxb, 256, Wkb, bk, kb, 256, 256, 1.0f, 1);
    wmma_gemm<2><<<dim3(BM / 16, 256 / 64), 32, 0, stream>>>(voxb, 256, Wvb, bv, vT, 256, 256, 1.0f, M);

    flash_attn<<<B * 8 * (N / 16), 32, 0, stream>>>(qb, kb, vT, Xf, N, M);

    wmma_gemm<1><<<dim3(BN / 16, 256 / 64), 32, 0, stream>>>(Xf, 512, Wfb, bfp, d_out, 512, 256, 1.0f, 1);

    (void)in_sizes; (void)n_in; (void)out_size; (void)ws_size;
}
